// PropagationLayer_70729521430966
// MI455X (gfx1250) — matrix operand order repaired and checked by reading, not compiled, not run on hardware
//
#include <hip/hip_runtime.h>

// ---------------------------------------------------------------------------
// Problem constants (from reference): B=64, V=2048, K=128, H=256
// ---------------------------------------------------------------------------
#define BATCH   64
#define VDIM    2048
#define KDIM    128
#define HDIM    256
#define EROWS   4097            // 1 + 2*V rows in eigs
#define NROWS   (BATCH * VDIM)  // 131072 rows for batchnorm
#define BN_EPSF 1e-5f

// Unified LDS word pitch: 20 u32 words (= 40 bf16) per row; 80 B, 16B aligned.
#define LDW 20

typedef __attribute__((ext_vector_type(16))) __bf16 v16bf;
typedef __attribute__((ext_vector_type(8)))  float  v8f;

// ---------------------------------------------------------------------------
// bf16 helpers — native hardware converts (v_cvt_pk_bf16_f32)
// ---------------------------------------------------------------------------
__device__ __forceinline__ unsigned pack_bf16(float lo, float hi) {
    __bf16 a = (__bf16)lo;
    __bf16 b = (__bf16)hi;
    unsigned short ua, ub;
    __builtin_memcpy(&ua, &a, 2);
    __builtin_memcpy(&ub, &b, 2);
    return (unsigned)ua | ((unsigned)ub << 16);
}
__device__ __forceinline__ float from_bf16(__bf16 b) { return (float)b; }

// 2x2 16-bit transpose via v_perm_b32: given p0 = [a0|a1], p1 = [b0|b1]
// produce w0 = [a0|b0], w1 = [a1|b1]  (lo|hi<<16)
__device__ __forceinline__ void interleave16(unsigned p0, unsigned p1,
                                             unsigned& w0, unsigned& w1) {
    w0 = __builtin_amdgcn_perm(p1, p0, 0x05040100u);  // bytes: p0.b01, p1.b01
    w1 = __builtin_amdgcn_perm(p1, p0, 0x07060302u);  // bytes: p0.b23, p1.b23
}

// ---------------------------------------------------------------------------
// WMMA: D(16x16 f32) = A(16x32 bf16) * B(32x16 bf16) + C
// ---------------------------------------------------------------------------
__device__ __forceinline__ v8f wmma_bf16(v16bf a, v16bf b, v8f c) {
    return __builtin_amdgcn_wmma_f32_16x16x32_bf16(
        /*neg_a=*/false, a, /*neg_b=*/false, b,
        /*c_mod=*/(short)0, c, /*reuse_a=*/false, /*reuse_b=*/false);
}

// ---------------------------------------------------------------------------
// LDS layouts (word = u32 packing bf16 elements k=2j [lo], k=2j+1 [hi]):
//   A tile: word (m, j)  at lds_a[m*LDW + j]   (m = output row, j = k-pair)
//   B tile: word (n, j)  at lds_b[n*LDW + j]   (n = output col, j = k-pair)
// Per-lane fragment words are contiguous -> each fragment is 2x ds_load_b128.
//
// ISA A-fragment (16-bit A 16x32): lane m=lane&15, half=lane>>4;
//   VGPR g: k0 = (g&4?16:0) + (half?8:0) + 2*(g&3)  ->  j runs: [4*half .. +3],
//   then [8+4*half .. +3].
// ISA B-fragment (32x16): lane n=lane&15, half=lane>>4; same k order as A
//   ->  j runs: [4*half .. +3], then [8+4*half .. +3].
// ---------------------------------------------------------------------------
__device__ __forceinline__ v16bf load_a_frag(const unsigned* p) {
    const int lane = threadIdx.x & 31;
    const int m = lane & 15, half = lane >> 4;
    union { uint4 q[2]; v16bf v; } f;
    f.q[0] = *(const uint4*)&p[m * LDW + 4 * half];
    f.q[1] = *(const uint4*)&p[m * LDW + 8 + 4 * half];
    return f.v;
}
__device__ __forceinline__ v16bf load_b_frag(const unsigned* p) {
    const int lane = threadIdx.x & 31;
    const int n = lane & 15, half = lane >> 4;
    union { uint4 q[2]; v16bf v; } f;
    f.q[0] = *(const uint4*)&p[n * LDW + 4 * half];
    f.q[1] = *(const uint4*)&p[n * LDW + 8 + 4 * half];
    return f.v;
}

// ---------------------------------------------------------------------------
// K0: zero the BN statistics buffer
// ---------------------------------------------------------------------------
__global__ __launch_bounds__(256) void zero_kernel(float* p, int n) {
    int i = blockIdx.x * 256 + threadIdx.x;
    if (i < n) p[i] = 0.0f;
}

// ---------------------------------------------------------------------------
// K1: hps[b,k,h] = band(k,h)*prop(k,h) * sum_v Einv[b,v,k]*h[b,v,h]
// Grid: (BATCH), Block 512 (16 waves). wr=w>>1 -> k rows 16wr.., wc=w&1 -> H
// half; 8 N-tiles per wave. Inner V in steps of 32.
// ---------------------------------------------------------------------------
__global__ __launch_bounds__(512)
void spectral_kernel(const float* __restrict__ hbuf, const float* __restrict__ eigs,
                     const float* __restrict__ ptime, const float* __restrict__ bmean,
                     const float* __restrict__ bstd, float* __restrict__ hps) {
    const int b    = blockIdx.x;
    const int t    = threadIdx.x;
    const int w    = t >> 5;
    const int lane = t & 31;
    const int nL   = lane & 15, half = lane >> 4;
    const int wr   = w >> 1, wc = w & 1;

    __shared__ __align__(16) unsigned lds_a[128 * LDW];  // A = Einv^T: [m(k)][j]
    __shared__ __align__(16) unsigned lds_b[256 * LDW];  // B = h:      [n(h)][j]

    const v8f vz = {0.f,0.f,0.f,0.f,0.f,0.f,0.f,0.f};
    v8f acc[8];
#pragma unroll
    for (int i = 0; i < 8; ++i) acc[i] = vz;

    const float* einv_base = eigs + (size_t)b * EROWS * KDIM + (size_t)(1 + VDIM) * KDIM;
    const float* h_base    = hbuf + (size_t)b * VDIM * HDIM;

    const int jp = t >> 5;           // k-pair index 0..15
    const int mb = (t & 31) * 4;     // A column (m) base: 128 / 32 thr
    const int nb = (t & 31) * 8;     // B column (n) base: 256 / 32 thr

    for (int v0 = 0; v0 < VDIM; v0 += 32) {
        // stage A: word(m=mb+i, j=jp) = pack(Einv[v0+2jp][m], Einv[v0+2jp+1][m])
        {
            const float4* p0 = (const float4*)(einv_base + (size_t)(v0 + 2*jp)     * KDIM + mb);
            const float4* p1 = (const float4*)(einv_base + (size_t)(v0 + 2*jp + 1) * KDIM + mb);
            float4 a0 = p0[0], a1 = p1[0];
            unsigned w0, w1, w2, w3;
            interleave16(pack_bf16(a0.x, a0.y), pack_bf16(a1.x, a1.y), w0, w1);
            interleave16(pack_bf16(a0.z, a0.w), pack_bf16(a1.z, a1.w), w2, w3);
            lds_a[(mb + 0) * LDW + jp] = w0;
            lds_a[(mb + 1) * LDW + jp] = w1;
            lds_a[(mb + 2) * LDW + jp] = w2;
            lds_a[(mb + 3) * LDW + jp] = w3;
        }
        // stage B: word(n=nb+i, j=jp) = pack(h[v0+2jp][n], h[v0+2jp+1][n])
        {
            const float4* p0 = (const float4*)(h_base + (size_t)(v0 + 2*jp)     * HDIM + nb);
            const float4* p1 = (const float4*)(h_base + (size_t)(v0 + 2*jp + 1) * HDIM + nb);
#pragma unroll
            for (int q = 0; q < 2; ++q) {
                float4 b0 = p0[q], b1 = p1[q];
                unsigned w0, w1, w2, w3;
                interleave16(pack_bf16(b0.x, b0.y), pack_bf16(b1.x, b1.y), w0, w1);
                interleave16(pack_bf16(b0.z, b0.w), pack_bf16(b1.z, b1.w), w2, w3);
                lds_b[(nb + q*4 + 0) * LDW + jp] = w0;
                lds_b[(nb + q*4 + 1) * LDW + jp] = w1;
                lds_b[(nb + q*4 + 2) * LDW + jp] = w2;
                lds_b[(nb + q*4 + 3) * LDW + jp] = w3;
            }
        }
        if (v0 + 32 < VDIM) {
            __builtin_prefetch(einv_base + (size_t)(v0 + 32 + 2*jp) * KDIM + mb, 0, 1);
            __builtin_prefetch(h_base + (size_t)(v0 + 32 + 2*jp) * HDIM + nb, 0, 1);
        }
        __syncthreads();
        v16bf afrag = load_a_frag(lds_a + 16 * wr * LDW);
#pragma unroll
        for (int tt = 0; tt < 8; ++tt) {
            v16bf bfrag = load_b_frag(lds_b + (128 * wc + 16 * tt) * LDW);
            acc[tt] = wmma_bf16(afrag, bfrag, acc[tt]);
        }
        __syncthreads();
    }

    // Epilogue: apply band * propagation scaling, store f32
    const float* ev_base = eigs + (size_t)b * EROWS * KDIM;  // row 0 = eigenvalues
#pragma unroll
    for (int tt = 0; tt < 8; ++tt) {
        const int hc = 128 * wc + 16 * tt + nL;
        const float pt = fmaxf(ptime[hc], 1e-6f);
        const float bm = fminf(fmaxf(bmean[hc], 0.001f), 5.0f);
        const float bs = fmaxf(bstd[hc], 1.0f);
        const float inv2s2 = 1.0f / (2.0f * bs * bs);
#pragma unroll
        for (int r = 0; r < 8; ++r) {
            const int k = 16 * wr + r + 8 * half;
            const float ev = ev_base[k];
            const float d  = bm - ev;
            const float sc = __expf(-d * d * inv2s2) * __expf(-ev * pt);
            hps[((size_t)(b * KDIM + k)) * HDIM + hc] = acc[tt][r] * sc;
        }
    }
}

// ---------------------------------------------------------------------------
// Staging helpers for the row-major GEMMs (128 rows x 32 k per step)
// ---------------------------------------------------------------------------
// A: pack 8 consecutive f32 (row-contiguous inner-k) into 4 contiguous words
__device__ __forceinline__ void stage_a_row(unsigned* lds_a, const float* src,
                                            int m, int jb /*word base*/) {
    const float4* p = (const float4*)src;
    float4 f0 = p[0], f1 = p[1];
    uint4 wv;
    wv.x = pack_bf16(f0.x, f0.y); wv.y = pack_bf16(f0.z, f0.w);
    wv.z = pack_bf16(f1.x, f1.y); wv.w = pack_bf16(f1.z, f1.w);
    *(uint4*)&lds_a[m * LDW + jb] = wv;
}
// B: word(n, j=jp) = pack(src_row0[n], src_row1[n]) for 8 consecutive n
__device__ __forceinline__ void stage_b_rows(unsigned* lds_b, const float* r0,
                                             const float* r1, int jp, int nb) {
    const float4* p0 = (const float4*)r0;
    const float4* p1 = (const float4*)r1;
#pragma unroll
    for (int q = 0; q < 2; ++q) {
        float4 b0 = p0[q], b1 = p1[q];
        unsigned w0, w1, w2, w3;
        interleave16(pack_bf16(b0.x, b0.y), pack_bf16(b1.x, b1.y), w0, w1);
        interleave16(pack_bf16(b0.z, b0.w), pack_bf16(b1.z, b1.w), w2, w3);
        lds_b[(nb + q*4 + 0) * LDW + jp] = w0;
        lds_b[(nb + q*4 + 1) * LDW + jp] = w1;
        lds_b[(nb + q*4 + 2) * LDW + jp] = w2;
        lds_b[(nb + q*4 + 3) * LDW + jp] = w3;
    }
}

// ---------------------------------------------------------------------------
// K2: M[b] = hps[b] (128x256 f32) @ w1b (256x256) -> bf16 (128x256)
// ---------------------------------------------------------------------------
__global__ __launch_bounds__(512)
void mgemm_kernel(const float* __restrict__ hps, const float* __restrict__ w1,
                  __bf16* __restrict__ Mb) {
    const int b    = blockIdx.x;
    const int t    = threadIdx.x;
    const int w    = t >> 5;
    const int lane = t & 31;
    const int nL   = lane & 15, half = lane >> 4;
    const int wr   = w >> 1, wc = w & 1;

    __shared__ __align__(16) unsigned lds_a[128 * LDW];
    __shared__ __align__(16) unsigned lds_b[256 * LDW];

    const v8f vz = {0.f,0.f,0.f,0.f,0.f,0.f,0.f,0.f};
    v8f acc[8];
#pragma unroll
    for (int i = 0; i < 8; ++i) acc[i] = vz;

    const int ma = t >> 2, ja = (t & 3) * 4;    // A: row 0..127, word base
    const int jp = t >> 5, nb = (t & 31) * 8;   // B: k-pair, col base

    for (int s = 0; s < 8; ++s) {
        const int c0 = s * 32;
        stage_a_row(lds_a, hps + ((size_t)(b * KDIM + ma)) * HDIM + c0 + ja * 2, ma, ja);
        stage_b_rows(lds_b,
                     w1 + (size_t)(HDIM + c0 + 2*jp)     * HDIM + nb,
                     w1 + (size_t)(HDIM + c0 + 2*jp + 1) * HDIM + nb, jp, nb);
        __syncthreads();
        v16bf afrag = load_a_frag(lds_a + 16 * wr * LDW);
#pragma unroll
        for (int tt = 0; tt < 8; ++tt) {
            v16bf bfrag = load_b_frag(lds_b + (128 * wc + 16 * tt) * LDW);
            acc[tt] = wmma_bf16(afrag, bfrag, acc[tt]);
        }
        __syncthreads();
    }
#pragma unroll
    for (int tt = 0; tt < 8; ++tt) {
        const int n = 128 * wc + 16 * tt + nL;
#pragma unroll
        for (int r = 0; r < 8; ++r) {
            const int k = 16 * wr + r + 8 * half;
            Mb[((size_t)(b * KDIM + k)) * HDIM + n] = (__bf16)acc[tt][r];
        }
    }
}

// ---------------------------------------------------------------------------
// K3: y1 = h @ w1a + E @ M[b] + b1  (fused inner 256+128), bf16 out
// Grid: (V/128, BATCH), Block 512.
// ---------------------------------------------------------------------------
__global__ __launch_bounds__(512)
void y1_kernel(const float* __restrict__ hbuf, const float* __restrict__ eigs,
               const float* __restrict__ w1, const __bf16* __restrict__ Mb,
               const float* __restrict__ b1, __bf16* __restrict__ y1) {
    const int b    = blockIdx.y;
    const int v0   = blockIdx.x * 128;
    const int t    = threadIdx.x;
    const int w    = t >> 5;
    const int lane = t & 31;
    const int nL   = lane & 15, half = lane >> 4;
    const int wr   = w >> 1, wc = w & 1;

    __shared__ __align__(16) unsigned lds_a[128 * LDW];
    __shared__ __align__(16) unsigned lds_b[256 * LDW];

    const v8f vz = {0.f,0.f,0.f,0.f,0.f,0.f,0.f,0.f};
    v8f acc[8];
#pragma unroll
    for (int i = 0; i < 8; ++i) acc[i] = vz;

    const int ma = t >> 2, ja = (t & 3) * 4;
    const int jp = t >> 5, nb = (t & 31) * 8;

    // ---- phase 1: inner over H (h @ w1a) ----
    for (int s = 0; s < 8; ++s) {
        const int c0 = s * 32;
        stage_a_row(lds_a, hbuf + ((size_t)(b * VDIM + v0 + ma)) * HDIM + c0 + ja * 2, ma, ja);
        stage_b_rows(lds_b,
                     w1 + (size_t)(c0 + 2*jp)     * HDIM + nb,
                     w1 + (size_t)(c0 + 2*jp + 1) * HDIM + nb, jp, nb);
        __syncthreads();
        v16bf afrag = load_a_frag(lds_a + 16 * wr * LDW);
#pragma unroll
        for (int tt = 0; tt < 8; ++tt) {
            v16bf bfrag = load_b_frag(lds_b + (128 * wc + 16 * tt) * LDW);
            acc[tt] = wmma_bf16(afrag, bfrag, acc[tt]);
        }
        __syncthreads();
    }
    // ---- phase 2: inner over K (E @ M[b]) ----
    const float* e_base = eigs + (size_t)b * EROWS * KDIM + (size_t)1 * KDIM;  // eigen_vecs
    for (int s = 0; s < 4; ++s) {
        const int c0 = s * 32;
        stage_a_row(lds_a, e_base + (size_t)(v0 + ma) * KDIM + c0 + ja * 2, ma, ja);
        {   // B already bf16: interleave rows 2jp / 2jp+1 into packed words
            const uint4 ra = *(const uint4*)(Mb + ((size_t)(b * KDIM + c0 + 2*jp))     * HDIM + nb);
            const uint4 rb = *(const uint4*)(Mb + ((size_t)(b * KDIM + c0 + 2*jp + 1)) * HDIM + nb);
            const unsigned wa[4] = {ra.x, ra.y, ra.z, ra.w};
            const unsigned wb[4] = {rb.x, rb.y, rb.z, rb.w};
#pragma unroll
            for (int q = 0; q < 4; ++q) {
                unsigned w0, w1;
                interleave16(wa[q], wb[q], w0, w1);
                lds_b[(nb + 2*q)     * LDW + jp] = w0;
                lds_b[(nb + 2*q + 1) * LDW + jp] = w1;
            }
        }
        __syncthreads();
        v16bf afrag = load_a_frag(lds_a + 16 * wr * LDW);
#pragma unroll
        for (int tt = 0; tt < 8; ++tt) {
            v16bf bfrag = load_b_frag(lds_b + (128 * wc + 16 * tt) * LDW);
            acc[tt] = wmma_bf16(afrag, bfrag, acc[tt]);
        }
        __syncthreads();
    }
    // epilogue: +b1, store bf16
#pragma unroll
    for (int tt = 0; tt < 8; ++tt) {
        const int n = 128 * wc + 16 * tt + nL;
        const float bias = b1[n];
#pragma unroll
        for (int r = 0; r < 8; ++r) {
            const int row = v0 + 16 * wr + r + 8 * half;
            y1[((size_t)(b * VDIM + row)) * HDIM + n] = (__bf16)(acc[tt][r] + bias);
        }
    }
}

// ---------------------------------------------------------------------------
// K4/K6: per-column sum & sum-of-squares for batchnorm (atomic accumulate)
// ---------------------------------------------------------------------------
__global__ __launch_bounds__(256)
void stats_kernel(const __bf16* __restrict__ x, float* __restrict__ sums, int rows_per_block) {
    const int c = threadIdx.x;           // column 0..255
    const size_t base = (size_t)blockIdx.x * rows_per_block * HDIM;
    float s = 0.f, q = 0.f;
    for (int r = 0; r < rows_per_block; ++r) {
        const float v = from_bf16(x[base + (size_t)r * HDIM + c]);
        s += v; q += v * v;
    }
    atomicAdd(&sums[c], s);
    atomicAdd(&sums[HDIM + c], q);
}

// ---------------------------------------------------------------------------
// K5: y2 = silu(bn1(y1)) @ w2 + b2   (BN+SiLU fused into A staging), bf16 out
// ---------------------------------------------------------------------------
__global__ __launch_bounds__(512)
void y2_kernel(const __bf16* __restrict__ y1, const float* __restrict__ w2,
               const float* __restrict__ stats1, const float* __restrict__ g1,
               const float* __restrict__ beta1, const float* __restrict__ b2,
               __bf16* __restrict__ y2) {
    const int b    = blockIdx.y;
    const int v0   = blockIdx.x * 128;
    const int t    = threadIdx.x;
    const int w    = t >> 5;
    const int lane = t & 31;
    const int nL   = lane & 15, half = lane >> 4;
    const int wr   = w >> 1, wc = w & 1;

    __shared__ __align__(16) unsigned lds_a[128 * LDW];
    __shared__ __align__(16) unsigned lds_b[256 * LDW];

    const v8f vz = {0.f,0.f,0.f,0.f,0.f,0.f,0.f,0.f};
    v8f acc[8];
#pragma unroll
    for (int i = 0; i < 8; ++i) acc[i] = vz;

    const float invN = 1.0f / (float)NROWS;
    const int ma = t >> 2, ja = (t & 3) * 4;
    const int jp = t >> 5, nb = (t & 31) * 8;

    for (int s = 0; s < 8; ++s) {
        const int c0 = s * 32;
        {   // A: bn1 + silu applied on the fly, pack pairs
            const size_t rbase = ((size_t)(b * VDIM + v0 + ma)) * HDIM;
            unsigned out[4];
#pragma unroll
            for (int q = 0; q < 4; ++q) {
                float sv2[2];
#pragma unroll
                for (int e = 0; e < 2; ++e) {
                    const int c = c0 + ja * 2 + q * 2 + e;
                    const float mean = stats1[c] * invN;
                    const float var  = stats1[HDIM + c] * invN - mean * mean;
                    const float sc   = g1[c] * rsqrtf(var + BN_EPSF);
                    const float xv   = from_bf16(y1[rbase + c]);
                    const float yv   = (xv - mean) * sc + beta1[c];
                    sv2[e] = yv / (1.0f + __expf(-yv));   // SiLU
                }
                out[q] = pack_bf16(sv2[0], sv2[1]);
            }
            *(uint4*)&lds_a[ma * LDW + ja] = make_uint4(out[0], out[1], out[2], out[3]);
        }
        stage_b_rows(lds_b,
                     w2 + (size_t)(c0 + 2*jp)     * HDIM + nb,
                     w2 + (size_t)(c0 + 2*jp + 1) * HDIM + nb, jp, nb);
        __syncthreads();
        v16bf afrag = load_a_frag(lds_a + 16 * wr * LDW);
#pragma unroll
        for (int tt = 0; tt < 8; ++tt) {
            v16bf bfrag = load_b_frag(lds_b + (128 * wc + 16 * tt) * LDW);
            acc[tt] = wmma_bf16(afrag, bfrag, acc[tt]);
        }
        __syncthreads();
    }
#pragma unroll
    for (int tt = 0; tt < 8; ++tt) {
        const int n = 128 * wc + 16 * tt + nL;
        const float bias = b2[n];
#pragma unroll
        for (int r = 0; r < 8; ++r) {
            const int row = v0 + 16 * wr + r + 8 * half;
            y2[((size_t)(b * VDIM + row)) * HDIM + n] = (__bf16)(acc[tt][r] + bias);
        }
    }
}

// ---------------------------------------------------------------------------
// K7: out = h + bn2(y2)   (elementwise)
// ---------------------------------------------------------------------------
__global__ __launch_bounds__(256)
void final_kernel(const float* __restrict__ hbuf, const __bf16* __restrict__ y2,
                  const float* __restrict__ stats2, const float* __restrict__ g2,
                  const float* __restrict__ beta2, float* __restrict__ out) {
    const size_t idx = (size_t)blockIdx.x * 256 + threadIdx.x;
    const int c = (int)(idx & (HDIM - 1));
    const float invN = 1.0f / (float)NROWS;
    const float mean = stats2[c] * invN;
    const float var  = stats2[HDIM + c] * invN - mean * mean;
    const float sc   = g2[c] * rsqrtf(var + BN_EPSF);
    out[idx] = hbuf[idx] + (from_bf16(y2[idx]) - mean) * sc + beta2[c];
}

// ---------------------------------------------------------------------------
// Launcher
// ---------------------------------------------------------------------------
extern "C" void kernel_launch(void* const* d_in, const int* in_sizes, int n_in,
                              void* d_out, int out_size, void* d_ws, size_t ws_size,
                              hipStream_t stream) {
    const float* h     = (const float*)d_in[0];
    const float* eigs  = (const float*)d_in[1];
    const float* ptime = (const float*)d_in[2];
    const float* bmean = (const float*)d_in[3];
    const float* bstd  = (const float*)d_in[4];
    const float* w1    = (const float*)d_in[5];
    const float* b1    = (const float*)d_in[6];
    const float* g1    = (const float*)d_in[7];
    const float* beta1 = (const float*)d_in[8];
    const float* w2    = (const float*)d_in[9];
    const float* b2    = (const float*)d_in[10];
    const float* g2    = (const float*)d_in[11];
    const float* beta2 = (const float*)d_in[12];
    float* out = (float*)d_out;

    char* ws = (char*)d_ws;
    float*  hps   = (float*)ws;                                   //  8 MB f32
    __bf16* Mb    = (__bf16*)(ws + ((size_t)8  << 20));           //  4 MB bf16
    __bf16* y1    = (__bf16*)(ws + ((size_t)12 << 20));           // 64 MB bf16
    __bf16* y2    = (__bf16*)(ws + ((size_t)76 << 20));           // 64 MB bf16
    float*  stats = (float*)(ws + ((size_t)140 << 20));           // 4*H f32

    zero_kernel<<<dim3(4), 256, 0, stream>>>(stats, 4 * HDIM);
    spectral_kernel<<<dim3(BATCH), 512, 0, stream>>>(h, eigs, ptime, bmean, bstd, hps);
    mgemm_kernel<<<dim3(BATCH), 512, 0, stream>>>(hps, w1, Mb);
    y1_kernel<<<dim3(VDIM / 128, BATCH), 512, 0, stream>>>(h, eigs, w1, Mb, b1, y1);
    stats_kernel<<<dim3(128), 256, 0, stream>>>(y1, stats, NROWS / 128);
    y2_kernel<<<dim3(VDIM / 128, BATCH), 512, 0, stream>>>(y1, w2, stats, g1, beta1, b2, y2);
    stats_kernel<<<dim3(128), 256, 0, stream>>>(y2, stats + 2 * HDIM, NROWS / 128);
    final_kernel<<<dim3((NROWS * HDIM) / 256), 256, 0, stream>>>(h, y2, stats + 2 * HDIM, g2, beta2, out);
}
